// Decoder_19834158973221
// MI455X (gfx1250) — matrix operand order repaired and checked by reading, compiled, not run on hardware
//
#include <hip/hip_runtime.h>

// ---------------------------------------------------------------------------
// LSTM decoder rollout for MI455X (gfx1250, wave32, WMMA).
// BS=8192, IN=48, HID=256, LAT=64, OUT=128, CAT=448, G=4*OUT=512, T=50.
//
// gates_t = base + h_t @ W_comb^T      (t >= 1, since yi == h after step 0)
//   base   = b_ih+b_hh + z@W_ih[:, :64]^T + enc@W_ih[:,192:]^T   (precomputed)
//   W_comb = W_ih[:, 64:192] + W_hh                              (f16, in LDS)
// Step 0 handled by a scalar setup kernel (u0/h0/c0 path differs).
// v3: j-loop kept as a runtime loop (#pragma unroll 1) so the scheduler's
//     pipelining window stays small -> no spills, no vgpr-msb churn.
// ---------------------------------------------------------------------------

#define BS   8192
#define IN_  48
#define HID  256
#define LAT  64
#define OUT_ 128
#define CAT  448
#define G4   512
#define T_   50
#define PITCH 136   // LDS row pitch in halfs (272B, 16B aligned, bank-spread)
#define CSTRIDE 68  // per-lane c-state stride in floats (272B, bank-spread)

typedef __attribute__((ext_vector_type(16))) _Float16 v16h;
typedef __attribute__((ext_vector_type(8)))  _Float16 v8h;
typedef __attribute__((ext_vector_type(8)))  float    v8f;
typedef __attribute__((ext_vector_type(4)))  float    v4f;

__device__ __forceinline__ float sigm(float x) {
    return __builtin_amdgcn_rcpf(1.0f + __expf(-x));   // v_rcp_f32, not IEEE div
}
__device__ __forceinline__ float tanh_(float x) {
    float e = __expf(-2.0f * x);
    return (1.0f - e) * __builtin_amdgcn_rcpf(1.0f + e);
}

// base[] is stored in "fragment order" so the main kernel initializes each
// 16x16 accumulator tile with one coalesced v8f (32B) load per lane.
__device__ __forceinline__ size_t frag_index(int b, int g) {
    int btile = b >> 4;           // 0..511
    int nt    = g >> 4;           // 0..31
    int lane  = (((b >> 3) & 1) << 4) | (g & 15);
    int r     = b & 7;
    return ((size_t)(btile * 32 + nt) * 32 + lane) * 8 + r;
}

// A-frag (16x32 f16): lane holds row m=lane&15; k = (e/8)*16 + (lane>>4)*8 + e%8
__device__ __forceinline__ v16h load_afrag(const _Float16* hb, int lane, int kc) {
    int m  = lane & 15;
    int hi = lane >> 4;
    const _Float16* p = hb + m * PITCH + kc * 32 + hi * 8;
    v8h lo = *(const v8h*)(p);        // k = kc*32 + hi*8 + 0..7
    v8h hh = *(const v8h*)(p + 16);   // k = kc*32 + 16 + hi*8 + 0..7
    return __builtin_shufflevector(lo, hh, 0,1,2,3,4,5,6,7,8,9,10,11,12,13,14,15);
}

// B-frag (32x16 f16): lane holds col n = nt*16 + (lane&15); k = (e/8)*16 + (lane>>4)*8 + e%8
// sW holds W_comb[g][k] row-major (g-major) with pitch PITCH, so
// B[k][n] = W_comb[n][k] as required for gates = h @ W_comb^T.
__device__ __forceinline__ v16h load_bfrag(const _Float16* w, int lane, int nt, int kc) {
    int n  = nt * 16 + (lane & 15);
    int hi = lane >> 4;
    const _Float16* p = w + n * PITCH + kc * 32 + hi * 8;
    v8h lo = *(const v8h*)(p);
    v8h hh = *(const v8h*)(p + 16);
    return __builtin_shufflevector(lo, hh, 0,1,2,3,4,5,6,7,8,9,10,11,12,13,14,15);
}

// ------------------------------ setup kernels ------------------------------

// u0 = x @ W_u0^T + b_u0
__global__ void k_u0(const float* __restrict__ x, const float* __restrict__ W_u0,
                     const float* __restrict__ b_u0, float* __restrict__ u0) {
    __shared__ float sx[IN_];
    int b = blockIdx.x, o = threadIdx.x;
    if (o < IN_) sx[o] = x[b * IN_ + o];
    __syncthreads();
    float acc = b_u0[o];
    const float* w = W_u0 + o * IN_;
    #pragma unroll 8
    for (int j = 0; j < IN_; ++j) acc += sx[j] * w[j];
    u0[b * OUT_ + o] = acc;
}

// h0 = xi0 @ W_h1^T + b_h1 ; c0 = xi0 @ W_h2^T + b_h2 ; xi0 = [z,u0,enc]
__global__ void k_h0c0(const float* __restrict__ z, const float* __restrict__ u0,
                       const float* __restrict__ enc,
                       const float* __restrict__ W_h1, const float* __restrict__ b_h1,
                       const float* __restrict__ W_h2, const float* __restrict__ b_h2,
                       float* __restrict__ h0, float* __restrict__ c0) {
    __shared__ float xi[CAT];
    int b = blockIdx.x, o = threadIdx.x;
    for (int idx = o; idx < CAT; idx += OUT_) {
        float v;
        if (idx < LAT)             v = z[b * LAT + idx];
        else if (idx < LAT + OUT_) v = u0[b * OUT_ + (idx - LAT)];
        else                       v = enc[b * HID + (idx - LAT - OUT_)];
        xi[idx] = v;
    }
    __syncthreads();
    float h = b_h1[o], c = b_h2[o];
    const float* w1 = W_h1 + o * CAT;
    const float* w2 = W_h2 + o * CAT;
    #pragma unroll 8
    for (int j = 0; j < CAT; ++j) { h += xi[j] * w1[j]; c += xi[j] * w2[j]; }
    h0[b * OUT_ + o] = h;
    c0[b * OUT_ + o] = c;
}

// base[b][g] = b_ih+b_hh + z@W_ih[:, :64]^T + enc@W_ih[:,192:448]^T (frag layout)
__global__ void k_base(const float* __restrict__ z, const float* __restrict__ enc,
                       const float* __restrict__ W_ih,
                       const float* __restrict__ b_ih, const float* __restrict__ b_hh,
                       float* __restrict__ baseF) {
    __shared__ float sz[LAT];
    __shared__ float se[HID];
    int b = blockIdx.x, g = threadIdx.x;
    if (g < LAT)                 sz[g] = z[b * LAT + g];
    if (g >= 64 && g < 64 + HID) se[g - 64] = enc[b * HID + (g - 64)];
    __syncthreads();
    float acc = b_ih[g] + b_hh[g];
    const float* w = W_ih + g * CAT;
    #pragma unroll 8
    for (int j = 0; j < LAT; ++j) acc += sz[j] * w[j];
    #pragma unroll 8
    for (int j = 0; j < HID; ++j) acc += se[j] * w[LAT + OUT_ + j];
    baseF[frag_index(b, g)] = acc;
}

// Step 0 (yi=u0, h=h0): gates0 = base + u0@W_ih[:,64:192]^T + h0@W_hh^T,
// then LSTM update in place (hbuf <- h1, cbuf <- c1), write out[:,0,:].
__global__ void k_step0(const float* __restrict__ u0,
                        const float* __restrict__ W_ih, const float* __restrict__ W_hh,
                        const float* __restrict__ baseF,
                        float* __restrict__ hbuf, float* __restrict__ cbuf,
                        float* __restrict__ out) {
    __shared__ float su[OUT_];
    __shared__ float sh[OUT_];
    __shared__ float sg[G4];
    int b = blockIdx.x, g = threadIdx.x;
    if (g < OUT_)          su[g] = u0[b * OUT_ + g];
    else if (g < 2 * OUT_) sh[g - OUT_] = hbuf[b * OUT_ + (g - OUT_)];
    __syncthreads();
    float acc = baseF[frag_index(b, g)];
    const float* wi = W_ih + g * CAT + LAT;
    const float* wh = W_hh + g * OUT_;
    #pragma unroll 8
    for (int j = 0; j < OUT_; ++j) acc += su[j] * wi[j] + sh[j] * wh[j];
    sg[g] = acc;
    __syncthreads();
    if (g < OUT_) {
        float ig = sg[g], fg = sg[OUT_ + g], gg = sg[2 * OUT_ + g], og = sg[3 * OUT_ + g];
        float c = sigm(fg) * cbuf[b * OUT_ + g] + sigm(ig) * tanh_(gg);
        float h = sigm(og) * tanh_(c);
        cbuf[b * OUT_ + g] = c;
        hbuf[b * OUT_ + g] = h;
        out[(size_t)b * (T_ * OUT_) + g] = h;   // t = 0
    }
}

// W_comb (f16): W_ih[:,64:192] + W_hh, row-major [512][128]
__global__ void k_wcomb(const float* __restrict__ W_ih, const float* __restrict__ W_hh,
                        _Float16* __restrict__ Wc) {
    int idx = blockIdx.x * blockDim.x + threadIdx.x;   // 0..65535
    int g = idx >> 7, k = idx & 127;
    Wc[idx] = (_Float16)(W_ih[g * CAT + LAT + k] + W_hh[g * OUT_ + k]);
}

// ------------------------------- main kernel -------------------------------
// 64 blocks x 256 threads (8 waves). Wave w owns 16 batch rows.
// 49 uniform steps: acc <- base (v8f L2 loads); 128 x v_wmma_f32_16x16x32_f16
// per step; LSTM elementwise with c state in LDS; h -> LDS (f16) for the next
// step's A-frags. j-loop is runtime to bound the scheduler's live window.
__global__ void __launch_bounds__(256)
k_lstm_main(const float* __restrict__ baseF, const float* __restrict__ h_in,
            const float* __restrict__ c_in, const _Float16* __restrict__ Wc,
            float* __restrict__ out) {
    extern __shared__ char smem_raw[];
    _Float16* sW = (_Float16*)smem_raw;               // 512 * PITCH halfs
    _Float16* sH = sW + G4 * PITCH;                   // 8 waves * 16 * PITCH halfs
    float*    sC = (float*)(smem_raw + (size_t)(G4 * PITCH + 8 * 16 * PITCH) * 2);

    int tid  = threadIdx.x;
    int lane = tid & 31;
    int wave = tid >> 5;
    int btile = blockIdx.x * 8 + wave;                // 16-row tile index
    int grow  = btile * 16;                           // first global batch row
    int lo = lane & 15, hi = lane >> 4;

    // cooperative weight load into LDS (padded pitch)
    for (int idx = tid; idx < G4 * OUT_; idx += 256) {
        int g = idx >> 7, k = idx & 127;
        sW[g * PITCH + k] = Wc[idx];
    }
    // stage initial h (h1 from step 0) into this wave's LDS tile, as f16
    _Float16* sHw = sH + wave * (16 * PITCH);
    for (int idx = lane; idx < 16 * OUT_; idx += 32) {
        int rr = idx >> 7, k = idx & 127;
        sHw[rr * PITCH + k] = (_Float16)h_in[(grow + rr) * OUT_ + k];
    }
    // c state lives in a wave-private LDS strip (C-fragment element order):
    // cLane[j*8 + r]  <=>  c[row = grow + r + 8*hi][col = 16*j + lo]
    float* cLane = sC + (wave * 32 + lane) * CSTRIDE;
    #pragma unroll
    for (int j = 0; j < 8; ++j)
        #pragma unroll
        for (int r = 0; r < 8; ++r)
            cLane[j * 8 + r] = c_in[(grow + r + 8 * hi) * OUT_ + 16 * j + lo];
    __syncthreads();   // sW visible to all waves; sHw/cLane are wave-private

    // per-lane base pointer: tile (j + 8q) is 256 floats apart in frag order
    const float* basePtr = baseF + ((size_t)(btile * 32) * 32 + lane) * 8;
    float* outPtr = out + (size_t)(grow + 8 * hi) * (T_ * OUT_) + lo;

    for (int t = 1; t < T_; ++t) {
        // A-frags for all 4 K-chunks (h of previous step), reused across j
        v16h a[4];
        #pragma unroll
        for (int kc = 0; kc < 4; ++kc) a[kc] = load_afrag(sHw, lane, kc);

        #pragma unroll 1   // runtime loop: keeps VGPR pressure bounded
        for (int j = 0; j < 8; ++j) {          // gate column group: i,f,g,o tiles
            v8f acc[4];
            #pragma unroll
            for (int q = 0; q < 4; ++q)
                acc[q] = *(const v8f*)(basePtr + (j + 8 * q) * 256);
            #pragma unroll
            for (int kc = 0; kc < 4; ++kc) {
                #pragma unroll
                for (int q = 0; q < 4; ++q) {
                    v16h bfr = load_bfrag(sW, lane, j + 8 * q, kc);
                    acc[q] = __builtin_amdgcn_wmma_f32_16x16x32_f16(
                        false, a[kc], false, bfr, (short)0, acc[q], false, false);
                }
            }
            // LSTM elementwise on matching C-layout elements; c from/to LDS
            v4f c0 = *(const v4f*)(cLane + j * 8);
            v4f c1 = *(const v4f*)(cLane + j * 8 + 4);
            v4f n0, n1;
            #pragma unroll
            for (int r = 0; r < 8; ++r) {
                float ig = acc[0][r], fg = acc[1][r], gg = acc[2][r], og = acc[3][r];
                float cold = (r < 4) ? c0[r] : c1[r - 4];
                float cc = sigm(fg) * cold + sigm(ig) * tanh_(gg);
                float hh = sigm(og) * tanh_(cc);
                if (r < 4) n0[r] = cc; else n1[r - 4] = cc;
                __builtin_nontemporal_store(
                    hh, outPtr + (size_t)r * (T_ * OUT_) + t * OUT_ + 16 * j);
                sHw[(r + 8 * hi) * PITCH + 16 * j + lo] = (_Float16)hh;
            }
            *(v4f*)(cLane + j * 8)     = n0;
            *(v4f*)(cLane + j * 8 + 4) = n1;
        }
        // same-wave LDS RAW (stores above -> next iter's A-frag/c loads) is
        // ordered by DScnt waits the compiler inserts; no barrier needed.
    }
}

// ------------------------------ host launcher ------------------------------

extern "C" void kernel_launch(void* const* d_in, const int* in_sizes, int n_in,
                              void* d_out, int out_size, void* d_ws, size_t ws_size,
                              hipStream_t stream) {
    const float* x    = (const float*)d_in[0];
    const float* enc  = (const float*)d_in[1];
    const float* z    = (const float*)d_in[2];
    const float* W_ih = (const float*)d_in[3];
    const float* W_hh = (const float*)d_in[4];
    const float* b_ih = (const float*)d_in[5];
    const float* b_hh = (const float*)d_in[6];
    const float* W_u0 = (const float*)d_in[7];
    const float* b_u0 = (const float*)d_in[8];
    const float* W_h1 = (const float*)d_in[9];
    const float* b_h1 = (const float*)d_in[10];
    const float* W_h2 = (const float*)d_in[11];
    const float* b_h2 = (const float*)d_in[12];
    float* out = (float*)d_out;

    // workspace layout (floats)
    float* ws    = (float*)d_ws;
    float* u0    = ws;                         // 8192*128
    float* hbuf  = ws + 1048576;               // 8192*128 (h0 -> h1)
    float* cbuf  = ws + 2097152;               // 8192*128 (c0 -> c1)
    float* baseF = ws + 3145728;               // 8192*512 in fragment order
    _Float16* Wc = (_Float16*)(ws + 3145728 + 4194304);  // 512*128 f16

    k_u0   <<<dim3(BS), dim3(OUT_), 0, stream>>>(x, W_u0, b_u0, u0);
    k_h0c0 <<<dim3(BS), dim3(OUT_), 0, stream>>>(z, u0, enc, W_h1, b_h1, W_h2, b_h2,
                                                 hbuf, cbuf);
    k_base <<<dim3(BS), dim3(G4), 0, stream>>>(z, enc, W_ih, b_ih, b_hh, baseF);
    k_step0<<<dim3(BS), dim3(G4), 0, stream>>>(u0, W_ih, W_hh, baseF, hbuf, cbuf, out);
    k_wcomb<<<dim3(256), dim3(256), 0, stream>>>(W_ih, W_hh, Wc);

    size_t shmem = (size_t)(G4 * PITCH + 8 * 16 * PITCH) * sizeof(_Float16)
                 + (size_t)(8 * 32 * CSTRIDE) * sizeof(float);   // 243,712 B
    k_lstm_main<<<dim3(BS / 128), dim3(256), shmem, stream>>>(baseF, hbuf, cbuf, Wc, out);
}
